// EncoderBlock_33595234189670
// MI455X (gfx1250) — matrix operand order repaired and checked by reading, not compiled, not run on hardware
//
#include <hip/hip_runtime.h>
#include <cstdint>
#include <cstddef>

typedef __bf16 bf16_t;
typedef bf16_t v16bf __attribute__((ext_vector_type(16)));
typedef bf16_t v8bf  __attribute__((ext_vector_type(8)));
typedef float  v8f   __attribute__((ext_vector_type(8)));
typedef uint32_t u32x4 __attribute__((ext_vector_type(4)));
typedef uint32_t u32x8 __attribute__((ext_vector_type(8)));

#define SEQ    1024
#define NB     4
#define NH     16
#define DK     64
#define DMODEL 1024
#define DFF    4096
#define NTOK   (NB*SEQ)

// ---- helpers -------------------------------------------------------------

static __device__ __forceinline__ v16bf combine8(v8bf lo, v8bf hi) {
  v16bf r;
#pragma unroll
  for (int i = 0; i < 8; ++i) { r[i] = lo[i]; r[i + 8] = hi[i]; }
  return r;
}

// ds_swizzle XOR broadcast within 16-lane half: imm = (xor<<10)|0x1f
#define SWZ(v, imm) __int_as_float(__builtin_amdgcn_ds_swizzle(__float_as_int(v), (imm)))

static __device__ __forceinline__ float half_reduce_max(float v) {
  v = fmaxf(v, SWZ(v, 0x041F));
  v = fmaxf(v, SWZ(v, 0x081F));
  v = fmaxf(v, SWZ(v, 0x101F));
  v = fmaxf(v, SWZ(v, 0x201F));
  return v;
}
static __device__ __forceinline__ float half_reduce_sum(float v) {
  v += SWZ(v, 0x041F);
  v += SWZ(v, 0x081F);
  v += SWZ(v, 0x101F);
  v += SWZ(v, 0x201F);
  return v;
}

// ---- TDM: 2-D tile DMA global->LDS (D# per CDNA5 ISA sec.8) -----
// 2-group descriptor (2-D tensor): group0 = 4 SGPRs, group1 = 8 SGPRs.
// tile_k contiguous elements per row, tile_rows rows, row stride = stride_el.
// LDS result: row-major, row pitch = tile_k elements (2B each).
static __device__ __forceinline__ void tdm_load_2d(uint32_t lds_off, uint64_t gaddr,
                                                   uint32_t tile_k, uint32_t tile_rows,
                                                   uint32_t stride_el) {
  u32x4 g0;
  g0[0] = 1u;                                                   // count=1 (user D#)
  g0[1] = lds_off;                                              // lds_addr [63:32]
  g0[2] = (uint32_t)gaddr;                                      // global_addr lo
  g0[3] = (uint32_t)((gaddr >> 32) & 0x01FFFFFFu) | (2u << 30); // addr hi | type=2
  u32x8 g1;
  g1[0] = (1u << 16);                                  // wg_mask=0, data_size=1 (2B)
  g1[1] = (tile_k & 0xFFFFu) << 16;                    // tensor_dim0[15:0]
  g1[2] = (tile_k >> 16) | ((tile_rows & 0xFFFFu) << 16);  // td0 hi | tensor_dim1 lo
  g1[3] = (tile_rows >> 16) | ((tile_k & 0xFFFFu) << 16);  // td1 hi | tile_dim0
  g1[4] = tile_rows & 0xFFFFu;                         // tile_dim1 | tile_dim2=0
  g1[5] = stride_el;                                   // tensor_dim0_stride lo
  g1[6] = 0u;                                          // stride0 hi | stride1 lo
  g1[7] = 0u;                                          // stride1 hi
  asm volatile("tensor_load_to_lds %0, %1" :: "s"(g0), "s"(g1) : "memory");
}

// ---- weight transpose + fp32->bf16 convert: WT[n][k] = W[k][n] -----------

__global__ __launch_bounds__(256)
void transpose_cvt_kernel(const float* __restrict__ W, bf16_t* __restrict__ WT,
                          int K, int N) {
  __shared__ float tile[32][33];
  const int tx = threadIdx.x, ty = threadIdx.y;
  const int n0 = blockIdx.x * 32, k0 = blockIdx.y * 32;
#pragma unroll
  for (int j = 0; j < 4; ++j) {
    int k = k0 + ty + j * 8;
    tile[ty + j * 8][tx] = W[(size_t)k * N + n0 + tx];
  }
  __syncthreads();
#pragma unroll
  for (int j = 0; j < 4; ++j) {
    int n = n0 + ty + j * 8;
    WT[(size_t)n * K + k0 + tx] = (bf16_t)tile[tx][ty + j * 8];
  }
}

// ---- LayerNorm (ddof=1, eps added to std), fp32 in -> bf16 out -----------

__global__ __launch_bounds__(256)
void layernorm_kernel(const float* __restrict__ x, const float* __restrict__ alpha,
                      const float* __restrict__ beta, bf16_t* __restrict__ out) {
  __shared__ float ssum[256], ssq[256];
  const int row = blockIdx.x;
  const int tid = threadIdx.x;
  const float* xr = x + (size_t)row * DMODEL;
  float s = 0.f, q = 0.f;
#pragma unroll
  for (int i = tid; i < DMODEL; i += 256) { float v = xr[i]; s += v; q += v * v; }
  ssum[tid] = s; ssq[tid] = q;
  __syncthreads();
  for (int st = 128; st > 0; st >>= 1) {
    if (tid < st) { ssum[tid] += ssum[tid + st]; ssq[tid] += ssq[tid + st]; }
    __syncthreads();
  }
  const float mean = ssum[0] * (1.0f / DMODEL);
  float var = (ssq[0] - (float)DMODEL * mean * mean) * (1.0f / (DMODEL - 1));
  var = fmaxf(var, 0.0f);
  const float sc = alpha[0] / (sqrtf(var) + 1e-6f);
  const float bt = beta[0];
  bf16_t* orow = out + (size_t)row * DMODEL;
#pragma unroll
  for (int i = tid; i < DMODEL; i += 256)
    orow[i] = (bf16_t)((xr[i] - mean) * sc + bt);
}

// ---- TDM-fed bf16 WMMA GEMM: C = A(MxK) * BT(NxK)^T + bias ---------------
// Block tile 64x128, 8 waves (2x4), wave tile 32x32, K staged 64 deep via
// tensor_load_to_lds with double-buffered LDS (A 8KB + B 16KB per buffer).
// MODE 0: -> bf16 row-major          MODE 1: relu -> bf16 row-major
// MODE 2: + residual -> fp32         MODE 3: -> bf16 per-head transposed V^T

#define KSTAGE 64
#define LDS_A_SZ (64 * KSTAGE)     // elements
#define LDS_B_SZ (128 * KSTAGE)    // elements

template <int MODE>
__global__ __launch_bounds__(256)
void gemm_bf16_wmma(const bf16_t* __restrict__ A, const bf16_t* __restrict__ BT,
                    const float* __restrict__ bias, const float* __restrict__ res,
                    float* __restrict__ outF, bf16_t* __restrict__ outB,
                    int M, int N, int K) {
  __shared__ bf16_t lds_a[2 * LDS_A_SZ];
  __shared__ bf16_t lds_b[2 * LDS_B_SZ];

  const int lane = threadIdx.x & 31;
  const int wid  = threadIdx.x >> 5;
  const int wm = wid >> 2, wn = wid & 3;
  const int rr = lane & 15, half = lane >> 4;
  const int m0 = blockIdx.y * 64 + wm * 32;
  const int n0 = blockIdx.x * 128 + wn * 32;
  const bool w0 = (wid == 0);

  const uint32_t lds_a_off = (uint32_t)(uintptr_t)(&lds_a[0]);  // LDS space offset
  const uint32_t lds_b_off = (uint32_t)(uintptr_t)(&lds_b[0]);
  const uint64_t ga_base = (uint64_t)(uintptr_t)(A + (size_t)(blockIdx.y * 64) * K);
  const uint64_t gb_base = (uint64_t)(uintptr_t)(BT + (size_t)(blockIdx.x * 128) * K);

  v8f acc[2][2];
#pragma unroll
  for (int a = 0; a < 2; ++a)
#pragma unroll
    for (int b = 0; b < 2; ++b) acc[a][b] = (v8f)0.0f;

  // prologue: stage 0 into buffer 0
  if (w0) {
    tdm_load_2d(lds_a_off, ga_base, KSTAGE, 64, (uint32_t)K);
    tdm_load_2d(lds_b_off, gb_base, KSTAGE, 128, (uint32_t)K);
  }

  int cur = 0;
  for (int ks = 0; ks < K; ks += KSTAGE) {
    if (w0) __builtin_amdgcn_s_wait_tensorcnt(0);  // current buffer landed
    __syncthreads();
    if (w0 && (ks + KSTAGE) < K) {                 // prefetch next stage
      const uint64_t kb = (uint64_t)(ks + KSTAGE) * 2u;
      const int nxt = cur ^ 1;
      tdm_load_2d(lds_a_off + nxt * LDS_A_SZ * 2, ga_base + kb, KSTAGE, 64, (uint32_t)K);
      tdm_load_2d(lds_b_off + nxt * LDS_B_SZ * 2, gb_base + kb, KSTAGE, 128, (uint32_t)K);
    }

    const bf16_t* la = lds_a + cur * LDS_A_SZ;
    const bf16_t* lb = lds_b + cur * LDS_B_SZ;
#pragma unroll
    for (int kk = 0; kk < KSTAGE; kk += 32) {
      v16bf Am[2], Bn[2];
#pragma unroll
      for (int mi = 0; mi < 2; ++mi) {
        const bf16_t* ap = la + (size_t)(wm * 32 + mi * 16 + rr) * KSTAGE + kk;
        Am[mi] = *(const v16bf*)(ap + half * 16);
      }
#pragma unroll
      for (int ni = 0; ni < 2; ++ni) {
        const bf16_t* bp = lb + (size_t)(wn * 32 + ni * 16 + rr) * KSTAGE + kk + half * 16;
        Bn[ni] = *(const v16bf*)bp;
      }
#pragma unroll
      for (int mi = 0; mi < 2; ++mi)
#pragma unroll
        for (int ni = 0; ni < 2; ++ni)
          acc[mi][ni] = __builtin_amdgcn_wmma_f32_16x16x32_bf16(
              false, Am[mi], false, Bn[ni], (short)0, acc[mi][ni], false, false);
    }
    cur ^= 1;
    __syncthreads();  // all reads of this buffer done before it is re-filled
  }

#pragma unroll
  for (int mi = 0; mi < 2; ++mi)
#pragma unroll
    for (int ni = 0; ni < 2; ++ni)
#pragma unroll
      for (int i = 0; i < 8; ++i) {
        const int gm = m0 + mi * 16 + half * 8 + i;
        const int gn = n0 + ni * 16 + rr;
        float v = acc[mi][ni][i] + bias[gn];
        if (MODE == 1) v = fmaxf(v, 0.0f);
        if (MODE == 2) {
          outF[(size_t)gm * N + gn] = res[(size_t)gm * N + gn] + v;
        } else if (MODE == 3) {
          const int b = gm >> 10, s = gm & 1023, h = gn >> 6, d = gn & 63;
          outB[((size_t)((b * NH + h) * DK + d)) * SEQ + s] = (bf16_t)v;
        } else {
          outB[(size_t)gm * N + gn] = (bf16_t)v;
        }
      }
}

// ---- flash attention: one wave = 16 query rows, online softmax -----------

__global__ __launch_bounds__(128)
void flash_attn_kernel(const bf16_t* __restrict__ q, const bf16_t* __restrict__ kmat,
                       const bf16_t* __restrict__ vT, const int* __restrict__ mask,
                       bf16_t* __restrict__ attn_out) {
  __shared__ bf16_t lds_p[4 * 16 * 64];
  const int lane = threadIdx.x & 31, wid = threadIdx.x >> 5;
  const int rr = lane & 15, half = lane >> 4;
  const int bh = blockIdx.y, b = bh >> 4, h = bh & 15;
  const int s0 = blockIdx.x * 64 + wid * 16;

  const bf16_t* qp = q + ((size_t)b * SEQ + s0) * DMODEL + h * DK;  // row stride DMODEL
  const bf16_t* kp = kmat + (size_t)b * SEQ * DMODEL + h * DK;      // row stride DMODEL
  const bf16_t* vp = vT + (size_t)bh * DK * SEQ;                    // [d][s]
  const int* mp = mask + (size_t)b * SEQ;
  bf16_t* lp = lds_p + wid * (16 * 64);

  v16bf Qa[2];
#pragma unroll
  for (int kf = 0; kf < 2; ++kf) {
    const bf16_t* ap = qp + (size_t)rr * DMODEL + kf * 32;
    Qa[kf] = *(const v16bf*)(ap + half * 16);
  }

  float mi_[8], li_[8];
  v8f O[4];
#pragma unroll
  for (int i = 0; i < 8; ++i) { mi_[i] = -1e30f; li_[i] = 0.f; }
#pragma unroll
  for (int g = 0; g < 4; ++g) O[g] = (v8f)0.0f;

  for (int kt = 0; kt < SEQ; kt += 64) {
    v8f S[4];
#pragma unroll
    for (int ng = 0; ng < 4; ++ng) {
      v8f s = (v8f)0.0f;
      const int key = kt + ng * 16 + rr;
#pragma unroll
      for (int kf = 0; kf < 2; ++kf) {
        v16bf Bv = *(const v16bf*)(kp + (size_t)key * DMODEL + kf * 32 + half * 16);
        s = __builtin_amdgcn_wmma_f32_16x16x32_bf16(false, Qa[kf], false, Bv,
                                                    (short)0, s, false, false);
      }
      const int mk = mp[key];
#pragma unroll
      for (int i = 0; i < 8; ++i) {
        float v = s[i] * 0.125f;  // 1/sqrt(64)
        if (mk == 0) v = -1e9f;
        s[i] = v;
      }
      S[ng] = s;
    }

#pragma unroll
    for (int i = 0; i < 8; ++i) {
      float mx = S[0][i];
      mx = fmaxf(mx, S[1][i]); mx = fmaxf(mx, S[2][i]); mx = fmaxf(mx, S[3][i]);
      mx = half_reduce_max(mx);
      const float mn = fmaxf(mi_[i], mx);
      const float corr = __expf(mi_[i] - mn);
      float rs = 0.f;
#pragma unroll
      for (int ng = 0; ng < 4; ++ng) {
        float p = __expf(S[ng][i] - mn);
        S[ng][i] = p;
        rs += p;
      }
      rs = half_reduce_sum(rs);
      li_[i] = li_[i] * corr + rs;
      mi_[i] = mn;
#pragma unroll
      for (int ng = 0; ng < 4; ++ng) O[ng][i] *= corr;
    }

    // C-layout -> LDS (row-major 16x64 bf16), then reload as A fragments
#pragma unroll
    for (int ng = 0; ng < 4; ++ng)
#pragma unroll
      for (int i = 0; i < 8; ++i)
        lp[(i + 8 * half) * 64 + ng * 16 + rr] = (bf16_t)S[ng][i];
    asm volatile("s_wait_dscnt 0" ::: "memory");

#pragma unroll
    for (int kf2 = 0; kf2 < 2; ++kf2) {
      const bf16_t* pp = lp + (size_t)rr * 64 + kf2 * 32;
      v16bf Pa = *(const v16bf*)(pp + half * 16);
#pragma unroll
      for (int ng = 0; ng < 4; ++ng) {
        v16bf Bv = *(const v16bf*)(vp + (size_t)(ng * 16 + rr) * SEQ + kt + kf2 * 32 + half * 16);
        O[ng] = __builtin_amdgcn_wmma_f32_16x16x32_bf16(false, Pa, false, Bv,
                                                        (short)0, O[ng], false, false);
      }
    }
  }

#pragma unroll
  for (int i = 0; i < 8; ++i) {
    const float inv = li_[i] > 0.f ? 1.0f / li_[i] : 0.f;
    const int s = s0 + i + 8 * half;
#pragma unroll
    for (int ng = 0; ng < 4; ++ng)
      attn_out[((size_t)b * SEQ + s) * DMODEL + h * DK + ng * 16 + rr] =
          (bf16_t)(O[ng][i] * inv);
  }
}

// ---- orchestration -------------------------------------------------------

extern "C" void kernel_launch(void* const* d_in, const int* in_sizes, int n_in,
                              void* d_out, int out_size, void* d_ws, size_t ws_size,
                              hipStream_t stream) {
  const float* x     = (const float*)d_in[0];
  const int*   mask  = (const int*)d_in[1];
  const float* Wq = (const float*)d_in[2];  const float* bq = (const float*)d_in[3];
  const float* Wk = (const float*)d_in[4];  const float* bk = (const float*)d_in[5];
  const float* Wv = (const float*)d_in[6];  const float* bv = (const float*)d_in[7];
  const float* Wo = (const float*)d_in[8];  const float* bo = (const float*)d_in[9];
  const float* W1 = (const float*)d_in[10]; const float* b1 = (const float*)d_in[11];
  const float* W2 = (const float*)d_in[12]; const float* b2 = (const float*)d_in[13];
  const float* alpha1 = (const float*)d_in[14]; const float* beta1 = (const float*)d_in[15];
  const float* alpha2 = (const float*)d_in[16]; const float* beta2 = (const float*)d_in[17];

  char* ws = (char*)d_ws;
  size_t off = 0;
  auto alloc = [&](size_t bytes) -> void* {
    void* p = ws + off;
    off += (bytes + 255) & ~(size_t)255;
    return p;
  };
  bf16_t* WqT = (bf16_t*)alloc((size_t)DMODEL * DMODEL * 2);
  bf16_t* WkT = (bf16_t*)alloc((size_t)DMODEL * DMODEL * 2);
  bf16_t* WvT = (bf16_t*)alloc((size_t)DMODEL * DMODEL * 2);
  bf16_t* WoT = (bf16_t*)alloc((size_t)DMODEL * DMODEL * 2);
  bf16_t* W1T = (bf16_t*)alloc((size_t)DFF * DMODEL * 2);
  bf16_t* W2T = (bf16_t*)alloc((size_t)DMODEL * DFF * 2);
  bf16_t* xn  = (bf16_t*)alloc((size_t)NTOK * DMODEL * 2);
  bf16_t* qb  = (bf16_t*)alloc((size_t)NTOK * DMODEL * 2);   // h1 aliases qb..attnb (32MB)
  bf16_t* kb  = (bf16_t*)alloc((size_t)NTOK * DMODEL * 2);
  bf16_t* vTb = (bf16_t*)alloc((size_t)NTOK * DMODEL * 2);
  bf16_t* attnb = (bf16_t*)alloc((size_t)NTOK * DMODEL * 2);
  float*  x2  = (float*)alloc((size_t)NTOK * DMODEL * 4);
  bf16_t* h1  = qb;  // 4096*4096*2 = 32MB fits in qb+kb+vTb+attnb
  (void)in_sizes; (void)n_in; (void)out_size; (void)ws_size;

  const dim3 tb(32, 8);
  transpose_cvt_kernel<<<dim3(DMODEL/32, DMODEL/32), tb, 0, stream>>>(Wq, WqT, DMODEL, DMODEL);
  transpose_cvt_kernel<<<dim3(DMODEL/32, DMODEL/32), tb, 0, stream>>>(Wk, WkT, DMODEL, DMODEL);
  transpose_cvt_kernel<<<dim3(DMODEL/32, DMODEL/32), tb, 0, stream>>>(Wv, WvT, DMODEL, DMODEL);
  transpose_cvt_kernel<<<dim3(DMODEL/32, DMODEL/32), tb, 0, stream>>>(Wo, WoT, DMODEL, DMODEL);
  transpose_cvt_kernel<<<dim3(DFF/32,    DMODEL/32), tb, 0, stream>>>(W1, W1T, DMODEL, DFF);
  transpose_cvt_kernel<<<dim3(DMODEL/32, DFF/32),    tb, 0, stream>>>(W2, W2T, DFF, DMODEL);

  // LN1
  layernorm_kernel<<<NTOK, 256, 0, stream>>>(x, alpha1, beta1, xn);

  // Q, K (bf16 row-major), V (per-head transposed)
  gemm_bf16_wmma<0><<<dim3(DMODEL/128, NTOK/64), 256, 0, stream>>>(
      xn, WqT, bq, nullptr, nullptr, qb, NTOK, DMODEL, DMODEL);
  gemm_bf16_wmma<0><<<dim3(DMODEL/128, NTOK/64), 256, 0, stream>>>(
      xn, WkT, bk, nullptr, nullptr, kb, NTOK, DMODEL, DMODEL);
  gemm_bf16_wmma<3><<<dim3(DMODEL/128, NTOK/64), 256, 0, stream>>>(
      xn, WvT, bv, nullptr, nullptr, vTb, NTOK, DMODEL, DMODEL);

  // attention
  flash_attn_kernel<<<dim3(SEQ/64, NB*NH), 128, 0, stream>>>(qb, kb, vTb, mask, attnb);

  // out-proj + residual -> x2 (fp32)
  gemm_bf16_wmma<2><<<dim3(DMODEL/128, NTOK/64), 256, 0, stream>>>(
      attnb, WoT, bo, x, x2, nullptr, NTOK, DMODEL, DMODEL);

  // LN2
  layernorm_kernel<<<NTOK, 256, 0, stream>>>(x2, alpha2, beta2, xn);

  // FFN1 (relu -> bf16)
  gemm_bf16_wmma<1><<<dim3(DFF/128, NTOK/64), 256, 0, stream>>>(
      xn, W1T, b1, nullptr, nullptr, h1, NTOK, DFF, DMODEL);

  // FFN2 + residual -> d_out (fp32)
  gemm_bf16_wmma<2><<<dim3(DMODEL/128, NTOK/64), 256, 0, stream>>>(
      h1, W2T, b2, x2, (float*)d_out, nullptr, NTOK, DMODEL, DFF);
}
